// ScaledDotProductAttentionResemble_69715909149280
// MI455X (gfx1250) — matrix operand order repaired and checked by reading, not compiled, not run on hardware
//
#include <hip/hip_runtime.h>
#include <hip/hip_bf16.h>

#define B_ 2
#define H_ 16
#define S_ 2048
#define D_ 64
#define NT_ (S_ / 16)   // 128 row/col tiles
#define NK_ (S_ / 32)   // 64 K-chunks of 32

typedef _Float16 half16 __attribute__((ext_vector_type(16)));
typedef _Float16 half8  __attribute__((ext_vector_type(8)));
typedef float    float8 __attribute__((ext_vector_type(8)));
typedef float    floatx4 __attribute__((ext_vector_type(4)));   // native vec for NT stores

// Workspace layout:
//   qpack : [B*H][NT_][2 kc][32 lane][16]  f16   (A fragments of q)
//   kpack : [B*H][NT_][2 kc][32 lane][16]  f16   (B fragments of k)
//   vpack : [B*H][NK_][4 nt][32 lane][16]  f16   (B fragments of v_res)
//   partials : 512 floats
#define QK_HALVES ((size_t)B_ * H_ * NT_ * 2 * 32 * 16)
#define V_HALVES  ((size_t)B_ * H_ * NK_ * 4 * 32 * 16)

// ---------------------------------------------------------------------------
// Kernel 1: sub_d + v_res packed into WMMA-B layout + loss partials.
// sub_d[b,s,d] = (sum_h (v[b,h,s,d] - mean_b v[.,h,s,d]))^2 / H^2
// ---------------------------------------------------------------------------
__global__ void pack_v_loss_kernel(const float* __restrict__ v,
                                   _Float16* __restrict__ vpack,
                                   float* __restrict__ partials) {
    const int idx = blockIdx.x * blockDim.x + threadIdx.x;   // 0 .. S_*D_-1
    const int s = idx / D_;
    const int d = idx % D_;

    float v0[H_], v1[H_];
    float acc0 = 0.f, acc1 = 0.f;
#pragma unroll
    for (int h = 0; h < H_; ++h) {
        float a = v[((size_t)(0 * H_ + h) * S_ + s) * D_ + d];
        float b = v[((size_t)(1 * H_ + h) * S_ + s) * D_ + d];
        v0[h] = a;
        v1[h] = b;
        float m = 0.5f * (a + b);          // batch mean (B=2)
        acc0 += a - m;
        acc1 += b - m;
    }
    const float inv_hh = 1.0f / (float)(H_ * H_);
    float sub0 = acc0 * acc0 * inv_hh;
    float sub1 = acc1 * acc1 * inv_hh;

    // packed B-fragment index for this (s,d)
    const int kt     = s >> 5;
    const int laneHi = (s >> 4) & 1;
    const int e      = s & 15;
    const int nt     = d >> 4;
    const int lane   = laneHi * 16 + (d & 15);
    const size_t idx0 = ((size_t)(kt * 4 + nt) * 32 + lane) * 16 + e;
    const size_t bhStride = (size_t)NK_ * 4 * 32 * 16;       // 131072 halves

#pragma unroll
    for (int h = 0; h < H_; ++h) {
        vpack[(size_t)(0 * H_ + h) * bhStride + idx0] = (_Float16)(v0[h] + sub0);
        vpack[(size_t)(1 * H_ + h) * bhStride + idx0] = (_Float16)(v1[h] + sub1);
    }

    __shared__ float red[256];
    red[threadIdx.x] = sub0 + sub1;
    __syncthreads();
    for (int off = 128; off > 0; off >>= 1) {
        if ((int)threadIdx.x < off) red[threadIdx.x] += red[threadIdx.x + off];
        __syncthreads();
    }
    if (threadIdx.x == 0) partials[blockIdx.x] = red[0];
}

// ---------------------------------------------------------------------------
// Kernel 2: pack q (A fragments) and k (B fragments) as f16, one pass.
// ---------------------------------------------------------------------------
__global__ void pack_qk_kernel(const float* __restrict__ q,
                               const float* __restrict__ k,
                               _Float16* __restrict__ qpack,
                               _Float16* __restrict__ kpack) {
    const int t = blockIdx.x * blockDim.x + threadIdx.x;  // ((bh*NT_+tile)*2+kc)*32+lane
    const int lane = t & 31;
    const int kc   = (t >> 5) & 1;
    const int tile = (t >> 6) & (NT_ - 1);
    const int bh   = t >> 13;

    const int M      = lane & 15;
    const int laneHi = lane >> 4;
    const float* qbh = q + (size_t)bh * S_ * D_;
    const float* kbh = k + (size_t)bh * D_ * S_;
    _Float16* qp = qpack + (size_t)t * 16;
    _Float16* kp = kpack + (size_t)t * 16;

    // A layout (16-bit 16x32): lane -> M; elem e -> K = ((e<8)?0:16)+laneHi*8+(e&7)
    const int kbase = laneHi * 8;
#pragma unroll
    for (int e = 0; e < 16; ++e) {
        int K = ((e < 8) ? 0 : 16) + kbase + (e & 7);
        qp[e] = (_Float16)qbh[(size_t)(tile * 16 + M) * D_ + kc * 32 + K];
    }
    // B layout (16-bit 32x16): lane -> N; elem e -> K = ((e<8)?0:16)+laneHi*8+(e&7)
    const int Kb = laneHi * 8;
#pragma unroll
    for (int e = 0; e < 16; ++e) {
        int d = kc * 32 + ((e < 8) ? 0 : 16) + Kb + (e & 7);
        kp[e] = (_Float16)kbh[(size_t)d * S_ + tile * 16 + M];
    }
}

// ---------------------------------------------------------------------------
// Kernel 3: reduce partials -> loss scalar (mean over B*S*1*D)
// ---------------------------------------------------------------------------
__global__ void loss_finalize_kernel(const float* __restrict__ partials, int n,
                                     float* __restrict__ lossOut) {
    __shared__ float red[256];
    float a = 0.f;
    for (int i = threadIdx.x; i < n; i += 256) a += partials[i];
    red[threadIdx.x] = a;
    __syncthreads();
    for (int off = 128; off > 0; off >>= 1) {
        if ((int)threadIdx.x < off) red[threadIdx.x] += red[threadIdx.x + off];
        __syncthreads();
    }
    if (threadIdx.x == 0) *lossOut = red[0] / (float)((size_t)B_ * S_ * D_);
}

// ---------------------------------------------------------------------------
// Kernel 4: causal attention, one 16-row tile of one (b,h) per workgroup.
// 256 threads = 8 wave32. LDS: 16 x S_ f16 score/prob strip (64 KB).
// ---------------------------------------------------------------------------
__global__ void attn_kernel(const _Float16* __restrict__ qpack,
                            const _Float16* __restrict__ kpack,
                            const _Float16* __restrict__ vpack,
                            const float* __restrict__ scalePtr,
                            float* __restrict__ outO,      // [B,H,S,D]
                            float* __restrict__ outP) {    // [B,H,S,S]
    extern __shared__ _Float16 scf[];                      // [16][S_] f16
    const int rt   = blockIdx.x % NT_;
    const int bh   = blockIdx.x / NT_;
    const int r0   = rt * 16;
    const int tid  = threadIdx.x;
    const int w    = tid >> 5;
    const int lane = tid & 31;
    const float scale = *scalePtr;
    const int ncol     = r0 + 16;                 // valid cols (multiple of 16)
    const int nk       = (ncol + 31) >> 5;        // K-chunks for PV
    const int ncol_pad = nk << 5;

    float* pOut = outP + (size_t)bh * S_ * S_ + (size_t)r0 * S_;
    float* oOut = outO + (size_t)bh * S_ * D_ + (size_t)r0 * D_;

    const int M      = lane & 15;
    const int laneHi = lane >> 4;

    // ---- Phase 1: QK^T into LDS (f16) ----
    const _Float16* qp = qpack + (((size_t)bh * NT_ + rt) * 2 * 32 + lane) * 16;
    const half16 aq0 = *(const half16*)(qp);
    const half16 aq1 = *(const half16*)(qp + 32 * 16);
    const _Float16* kbase_p = kpack + ((size_t)bh * NT_ * 2 * 32 + lane) * 16;

    // unmasked tiles jt < rt
    for (int jt = w; jt < rt; jt += 8) {
        const _Float16* kp = kbase_p + (size_t)jt * 2 * 32 * 16;
        // speculative prefetch of the tile 8 ahead, i.e. this wave's next iteration (always in-bounds
        // of the workspace allocation; dropped silently otherwise)
        __builtin_prefetch(kp + (size_t)8 * 2 * 32 * 16, 0, 0);
        half16 bk0 = *(const half16*)(kp);
        half16 bk1 = *(const half16*)(kp + 32 * 16);
        float8 c = {};
        c = __builtin_amdgcn_wmma_f32_16x16x32_f16(false, aq0, false, bk0,
                                                   (short)0, c, false, false);
        c = __builtin_amdgcn_wmma_f32_16x16x32_f16(false, aq1, false, bk1,
                                                   (short)0, c, false, false);
        const int j0 = jt * 16;
#pragma unroll
        for (int r = 0; r < 8; ++r) {
            int row = r + laneHi * 8;
            scf[row * S_ + j0 + M] = (_Float16)(c[r] * scale);
        }
    }
    // diagonal tile jt == rt (only tile needing the causal mask)
    if (w == (rt & 7)) {
        const _Float16* kp = kbase_p + (size_t)rt * 2 * 32 * 16;
        half16 bk0 = *(const half16*)(kp);
        half16 bk1 = *(const half16*)(kp + 32 * 16);
        float8 c = {};
        c = __builtin_amdgcn_wmma_f32_16x16x32_f16(false, aq0, false, bk0,
                                                   (short)0, c, false, false);
        c = __builtin_amdgcn_wmma_f32_16x16x32_f16(false, aq1, false, bk1,
                                                   (short)0, c, false, false);
#pragma unroll
        for (int r = 0; r < 8; ++r) {
            int row = r + laneHi * 8;              // in-tile row
            float val = c[r] * scale;
            if (M > row) val = -__builtin_inff();  // causal: col r0+M > row r0+row
            scf[row * S_ + r0 + M] = (_Float16)val;
        }
    }
    __syncthreads();

    // ---- Phase 2: softmax per row (fp32 math over f16 scores) ----
    for (int row = w; row < 16; row += 8) {
        _Float16* rp = scf + row * S_;
        const int nchunk = ncol >> 3;                      // half8 chunks
        float mx = -__builtin_inff();
        for (int cch = lane; cch < nchunk; cch += 32) {
            half8 hv = *(const half8*)(rp + cch * 8);
#pragma unroll
            for (int e2 = 0; e2 < 8; ++e2) mx = fmaxf(mx, (float)hv[e2]);
        }
#pragma unroll
        for (int off = 16; off > 0; off >>= 1) mx = fmaxf(mx, __shfl_xor(mx, off, 32));
        float sum = 0.f;
        for (int cch = lane; cch < nchunk; cch += 32) {
            half8 hv = *(const half8*)(rp + cch * 8);
#pragma unroll
            for (int e2 = 0; e2 < 8; ++e2) sum += __expf((float)hv[e2] - mx);
        }
#pragma unroll
        for (int off = 16; off > 0; off >>= 1) sum += __shfl_xor(sum, off, 32);
        const float inv = 1.0f / sum;
        float* gp = pOut + (size_t)row * S_;
        for (int cch = lane; cch < nchunk; cch += 32) {
            half8 hv = *(const half8*)(rp + cch * 8);
            float p[8];
            half8 ph;
#pragma unroll
            for (int e2 = 0; e2 < 8; ++e2) {
                p[e2] = __expf((float)hv[e2] - mx) * inv;
                ph[e2] = (_Float16)p[e2];
            }
            *(half8*)(rp + cch * 8) = ph;                  // keep p (f16) in LDS
            // streaming fp32 attn_weights: non-temporal (never re-read, 537 MB)
            floatx4 p0 = { p[0], p[1], p[2], p[3] };
            floatx4 p1 = { p[4], p[5], p[6], p[7] };
            __builtin_nontemporal_store(p0, (floatx4*)(gp + cch * 8));
            __builtin_nontemporal_store(p1, (floatx4*)(gp + cch * 8) + 1);
        }
        // zero-pad p to the 32-col chunk boundary (removes K guards in PV)
        for (int j = ncol + lane; j < ncol_pad; j += 32) rp[j] = (_Float16)0.f;
    }
    __syncthreads();

    // ---- Phase 3 ----
    if (w < 4) {
        // waves 0-3: output tile 16x64; wave w -> columns d0 = w*16
        float8 c = {};
        const _Float16* rowp = scf + M * S_;
        const int kbase = laneHi * 16;
        const _Float16* vp = vpack + ((size_t)bh * NK_ * 4 + w) * 32 * 16 + (size_t)lane * 16;
        for (int kt = 0; kt < nk; ++kt) {
            // A fragment: two contiguous 8-half runs per lane (ds_load_b128 x2)
            half8 lo = *(const half8*)(rowp + kt * 32 + kbase);
            half8 hi = *(const half8*)(rowp + kt * 32 + kbase + 8);
            half16 ap;
#pragma unroll
            for (int e2 = 0; e2 < 8; ++e2) { ap[e2] = lo[e2]; ap[8 + e2] = hi[e2]; }
            half16 bv = *(const half16*)(vp + (size_t)kt * 4 * 32 * 16);
            c = __builtin_amdgcn_wmma_f32_16x16x32_f16(false, ap, false, bv,
                                                       (short)0, c, false, false);
        }
#pragma unroll
        for (int r = 0; r < 8; ++r) {
            int row = r + laneHi * 8;
            oOut[(size_t)row * D_ + w * 16 + M] = c[r];
        }
    } else {
        // waves 4-7: zero-fill masked attn_weights columns [ncol, S_) (streaming)
        floatx4 z = { 0.f, 0.f, 0.f, 0.f };
        const int tailv = (S_ - ncol) >> 2;
        const int total = 16 * tailv;
        for (int i = tid - 128; i < total; i += 128) {
            int row = i / tailv;
            int c4  = i % tailv;
            __builtin_nontemporal_store(z, (floatx4*)(pOut + (size_t)row * S_ + ncol) + c4);
        }
    }
}

// ---------------------------------------------------------------------------
// Host launcher
// Inputs: d_in[0]=q [B,H,S,D] f32, d_in[1]=k [B,H,D,S] f32, d_in[2]=v f32,
//         d_in[3]=scale (1 elem), d_in[4]=attn_mask (causal, ignored)
// Output: [ output (B*H*S*D) | attn_weights (B*H*S*S) | loss (1) ] f32
// ---------------------------------------------------------------------------
extern "C" void kernel_launch(void* const* d_in, const int* in_sizes, int n_in,
                              void* d_out, int out_size, void* d_ws, size_t ws_size,
                              hipStream_t stream) {
    const float* q     = (const float*)d_in[0];
    const float* k     = (const float*)d_in[1];
    const float* v     = (const float*)d_in[2];
    const float* scale = (const float*)d_in[3];

    float* outO    = (float*)d_out;
    float* outP    = outO + (size_t)B_ * H_ * S_ * D_;
    float* outLoss = outP + (size_t)B_ * H_ * S_ * S_;

    _Float16* qpack = (_Float16*)d_ws;
    _Float16* kpack = qpack + QK_HALVES;
    _Float16* vpack = kpack + QK_HALVES;
    float* partials = (float*)(vpack + V_HALVES);

    const int nPart = (S_ * D_) / 256;                     // 512 blocks
    pack_v_loss_kernel<<<nPart, 256, 0, stream>>>(v, vpack, partials);
    pack_qk_kernel<<<(int)(QK_HALVES / 16 / 256), 256, 0, stream>>>(q, k, qpack, kpack);
    loss_finalize_kernel<<<1, 256, 0, stream>>>(partials, nPart, outLoss);

    const size_t ldsBytes = (size_t)16 * S_ * sizeof(_Float16);  // 64 KB
    (void)hipFuncSetAttribute((const void*)attn_kernel,
                              hipFuncAttributeMaxDynamicSharedMemorySize,
                              (int)ldsBytes);
    attn_kernel<<<B_ * H_ * NT_, 256, ldsBytes, stream>>>(
        qpack, kpack, vpack, scale, outO, outP);
}
